// SelfAttention_32203664785753
// MI455X (gfx1250) — matrix operand order repaired and checked by reading, not compiled, not run on hardware
//
#include <hip/hip_runtime.h>

// ---------------------------------------------------------------------------
// MI455X (gfx1250, wave32) fused multi-head self-attention, bf16 WMMA path.
// All matmuls use v_wmma_f32_16x16x32_bf16 (fp32 accumulate).
// Attention: 4 waves/WG share one (b,h); K/V tiles staged cooperatively in
// LDS (40KB/WG of the WGP's 320KB) cutting L2 traffic 4x.
// __launch_bounds__(128,2): 4-wave blocks, <=512 VGPRs/wave -> no spills.
// ---------------------------------------------------------------------------

typedef __attribute__((ext_vector_type(16))) __bf16 v16bf;
typedef __attribute__((ext_vector_type(8)))  float  v8f;

struct alignas(16) U4 { unsigned a, b, c, d; };
union V16 { v16bf v; U4 h[2]; };

static __device__ __forceinline__ __bf16 f2bf(float f) {
    return static_cast<__bf16>(f);   // native gfx1250 f32->bf16 cvt (RNE)
}

#define WMMA_BF16(A, B, C) \
    __builtin_amdgcn_wmma_f32_16x16x32_bf16(false, (A), false, (B), (short)0, (C), false, false)

static constexpr int Bn = 4, T = 2048, E = 128, H = 8;
static constexpr int NE = H * E;      // 1024
static constexpr int M  = Bn * T;     // 8192
static constexpr float QK_SCALE = 0.29730177875068026f;  // 128^(-1/4)

// ---------------------------------------------------------------------------
// Prep kernels: one-time f32 -> bf16 conversions (tiny vs. 77 GFLOP of GEMM).
// ---------------------------------------------------------------------------
__global__ void xconv_kernel(const float* __restrict__ x, __bf16* __restrict__ xb) {
    int idx = blockIdx.x * 256 + threadIdx.x;          // < 1048576
    xb[idx] = f2bf(x[idx]);
}

// Wq/Wk/Wv [128,1024] f32 -> WT[mat][1024,128] bf16 (transposed: contiguous in k)
__global__ void wconv_kernel(const float* __restrict__ Wq,
                             const float* __restrict__ Wk,
                             const float* __restrict__ Wv,
                             __bf16* __restrict__ WT) {
    int idx = blockIdx.x * 256 + threadIdx.x;          // < 393216
    int mat = idx >> 17;
    int r   = idx & 131071;
    int k = r >> 10, n = r & 1023;                     // W row-major [k][n]
    const float* W = (mat == 0) ? Wq : (mat == 1) ? Wk : Wv;
    WT[(size_t)mat * (NE * E) + (size_t)n * E + k] = f2bf(W[r]);
}

// Wu [1024,128] f32 -> WuT [128,1024] bf16
__global__ void wuconv_kernel(const float* __restrict__ Wu, __bf16* __restrict__ WuT) {
    int idx = blockIdx.x * 256 + threadIdx.x;          // < 131072
    int k = idx >> 7, e = idx & 127;
    WuT[(size_t)e * NE + k] = f2bf(Wu[idx]);
}

// ---------------------------------------------------------------------------
// Kernel 1: fused QKV projection.  Xb[8192,128] @ W[128,1024] for {Wq,Wk,Wv}.
// One wave per 32x64 output strip: 2 A-frags x 4 B-frags = 8 WMMAs per chunk,
// so each B-fragment is reused twice and each A-fragment 4 times.
// q,k scaled by 128^-1/4, stored [b,h,t,e] bf16.  v stored transposed [b,h,e,t].
// ---------------------------------------------------------------------------
__global__ __launch_bounds__(128, 2)
void qkv_kernel(const __bf16* __restrict__ xb,
                const __bf16* __restrict__ WT,
                __bf16* __restrict__ qb,
                __bf16* __restrict__ kb,
                __bf16* __restrict__ vT) {
    int wave = threadIdx.x >> 5, lane = threadIdx.x & 31;
    int job  = blockIdx.x * 4 + wave;       // 3 * 256 * 16 = 12288 jobs
    int mat  = job >> 12;                   // 0=Q,1=K,2=V
    int rem  = job & 4095;
    int m0 = (rem >> 4) * 32, n0 = (rem & 15) * 64;
    int lm = lane & 15, hi = lane >> 4, hi8 = hi * 8, hi16 = hi * 16;

    v8f acc[2][4];
    #pragma unroll
    for (int mi = 0; mi < 2; mi++)
        #pragma unroll
        for (int j = 0; j < 4; j++) acc[mi][j] = {};

    const __bf16* xr0 = xb + (size_t)(m0 + lm) * E;                         // A rows
    const __bf16* xr1 = xr0 + (size_t)16 * E;
    const __bf16* wr  = WT + (size_t)mat * (NE * E) + (size_t)(n0 + lm) * E; // B cols
    #pragma unroll
    for (int kc = 0; kc < 4; kc++) {
        V16 a0, a1;
        a0.h[0] = *(const U4*)(xr0 + kc * 32 + hi16);
        a0.h[1] = *(const U4*)(xr0 + kc * 32 + hi16 + 8);
        a1.h[0] = *(const U4*)(xr1 + kc * 32 + hi16);
        a1.h[1] = *(const U4*)(xr1 + kc * 32 + hi16 + 8);
        #pragma unroll
        for (int j = 0; j < 4; j++) {
            V16 w;
            const __bf16* wj = wr + (size_t)j * 16 * E + kc * 32 + hi16;
            w.h[0] = *(const U4*)wj;
            w.h[1] = *(const U4*)(wj + 8);
            acc[0][j] = WMMA_BF16(a0.v, w.v, acc[0][j]);
            acc[1][j] = WMMA_BF16(a1.v, w.v, acc[1][j]);
        }
    }

    float scale = (mat == 2) ? 1.0f : QK_SCALE;
    #pragma unroll
    for (int mi = 0; mi < 2; mi++) {
        #pragma unroll
        for (int j = 0; j < 4; j++) {
            int n = n0 + j * 16 + lm;
            int hh = n >> 7, ee = n & 127;
            #pragma unroll
            for (int r = 0; r < 8; r++) {
                int m = m0 + mi * 16 + hi8 + r;   // D: row = (lane<16 ? r : 8+r)
                int bb = m >> 11, tt = m & 2047;
                __bf16 val = f2bf(acc[mi][j][r] * scale);
                if (mat == 2)
                    vT[(((size_t)bb * H + hh) * E + ee) * T + tt] = val;
                else {
                    __bf16* dst = (mat == 0) ? qb : kb;
                    dst[(((size_t)bb * H + hh) * T + tt) * E + ee] = val;
                }
            }
        }
    }
}

// ---------------------------------------------------------------------------
// Kernel 2: flash attention.  Block = 4 waves sharing one (b,h); each wave
// owns 16 query rows.  Per 64-key chunk: K(16KB)+V(16KB) tiles staged
// cooperatively into LDS, then 16 score WMMAs + online softmax + 16 PV WMMAs
// per wave, all operand B-fragments read from LDS.
// ---------------------------------------------------------------------------
__global__ __launch_bounds__(128, 2)
void attn_kernel(const __bf16* __restrict__ q,
                 const __bf16* __restrict__ kmat,
                 const __bf16* __restrict__ vT,
                 const int*    __restrict__ mask,
                 __bf16*       __restrict__ aout) {
    __shared__ __bf16 kt[64 * 128];        // K tile  [key][e]      16KB
    __shared__ __bf16 vt[128 * 64];        // V tile  [e][key]      16KB
    __shared__ __bf16 plds[4][16 * 64];    // per-wave P staging     8KB
    int tid = threadIdx.x;
    int wave = tid >> 5, lane = tid & 31;
    int bh = blockIdx.x >> 5, qblk = blockIdx.x & 31;   // 32 bh * 32 qblocks
    int b = bh >> 3, h = bh & 7;
    int q0 = qblk * 64 + wave * 16;
    int lm = lane & 15, hi = lane >> 4, hi8 = hi * 8, hi16 = hi * 16;

    const __bf16* kbase = kmat + (size_t)bh * T * E;
    const __bf16* vbase = vT   + (size_t)bh * E * T;

    // Q A-fragments for all 4 K-chunks of e=128 (kept in registers)
    V16 aq[4];
    {
        const __bf16* qr = q + ((size_t)bh * T + q0 + lm) * E;
        #pragma unroll
        for (int kc = 0; kc < 4; kc++) {
            aq[kc].h[0] = *(const U4*)(qr + kc * 32 + hi16);
            aq[kc].h[1] = *(const U4*)(qr + kc * 32 + hi16 + 8);
        }
    }
    int qm[8];
    #pragma unroll
    for (int r = 0; r < 8; r++) qm[r] = mask[b * T + q0 + hi8 + r];

    float mrun[8], lrun[8];
    v8f O[8];
    #pragma unroll
    for (int r = 0; r < 8; r++) { mrun[r] = -3.0e38f; lrun[r] = 0.0f; }
    #pragma unroll
    for (int c = 0; c < 8; c++) O[c] = {};

    __bf16* pw = plds[wave];

    for (int n0 = 0; n0 < T; n0 += 64) {
        // ---- cooperative K/V tile staging (coalesced 16B chunks) --------
        #pragma unroll
        for (int i = 0; i < 8; i++) {
            int ck = i * 128 + tid;                    // 1024 chunks of 8 elems
            int krow = ck >> 4, kcol = (ck & 15) * 8;  // K: 64 rows x 128
            *(U4*)&kt[krow * 128 + kcol] =
                *(const U4*)(kbase + (size_t)(n0 + krow) * E + kcol);
            int ve = ck >> 3, vk = (ck & 7) * 8;       // V: 128 rows x 64
            *(U4*)&vt[ve * 64 + vk] =
                *(const U4*)(vbase + (size_t)ve * T + n0 + vk);
        }
        // prefetch next key tile rows (global_prefetch_b8)
        if (n0 + 64 < T) {
            __builtin_prefetch(kbase + (size_t)(n0 + 64 + (tid >> 1)) * E, 0, 1);
            __builtin_prefetch(vbase + (size_t)tid * T + n0 + 64, 0, 1);
        }
        __syncthreads();

        // ---- scores: S[16x64] = Q[16x128] @ K^T  (B-frags from LDS) -----
        v8f s[4];
        #pragma unroll
        for (int j = 0; j < 4; j++) {
            s[j] = {};
            #pragma unroll
            for (int kc = 0; kc < 4; kc++) {
                V16 bk;
                const __bf16* kr = &kt[(j * 16 + lm) * 128 + kc * 32 + hi16];
                bk.h[0] = *(const U4*)kr;
                bk.h[1] = *(const U4*)(kr + 8);
                s[j] = WMMA_BF16(aq[kc].v, bk.v, s[j]);
            }
        }

        // ---- masking + online softmax (row reductions over 16-lane groups)
        int km[4];
        #pragma unroll
        for (int j = 0; j < 4; j++) km[j] = mask[b * T + n0 + j * 16 + lm];
        #pragma unroll
        for (int r = 0; r < 8; r++) {
            #pragma unroll
            for (int j = 0; j < 4; j++)
                if (!(qm[r] && km[j])) s[j][r] = -1.0e9f;
            float t = fmaxf(fmaxf(s[0][r], s[1][r]), fmaxf(s[2][r], s[3][r]));
            #pragma unroll
            for (int d = 1; d < 16; d <<= 1) t = fmaxf(t, __shfl_xor(t, d, 32));
            float mn = fmaxf(mrun[r], t);
            float alpha = __expf(mrun[r] - mn);
            mrun[r] = mn;
            float p0 = __expf(s[0][r] - mn), p1 = __expf(s[1][r] - mn);
            float p2 = __expf(s[2][r] - mn), p3 = __expf(s[3][r] - mn);
            s[0][r] = p0; s[1][r] = p1; s[2][r] = p2; s[3][r] = p3;
            float rs = (p0 + p1) + (p2 + p3);
            #pragma unroll
            for (int d = 1; d < 16; d <<= 1) rs += __shfl_xor(rs, d, 32);
            lrun[r] = lrun[r] * alpha + rs;
            #pragma unroll
            for (int c = 0; c < 8; c++) O[c][r] *= alpha;
        }

        // ---- P: C-layout -> A-fragment layout via per-wave LDS tile -----
        #pragma unroll
        for (int r = 0; r < 8; r++) {
            #pragma unroll
            for (int j = 0; j < 4; j++)
                pw[(hi8 + r) * 64 + j * 16 + lm] = f2bf(s[j][r]);
        }
        V16 pa0, pa1;  // DS ops are in-order within a wave; compiler waits DScnt
        pa0.h[0] = *(const U4*)&pw[lm * 64 + hi16];
        pa0.h[1] = *(const U4*)&pw[lm * 64 + hi16 + 8];
        pa1.h[0] = *(const U4*)&pw[lm * 64 + 32 + hi16];
        pa1.h[1] = *(const U4*)&pw[lm * 64 + 32 + hi16 + 8];

        // ---- O[16x128] += P[16x64] @ V[64x128]  (B-frags from LDS) ------
        #pragma unroll
        for (int c = 0; c < 8; c++) {
            V16 bv0, bv1;
            const __bf16* vr = &vt[(c * 16 + lm) * 64];
            bv0.h[0] = *(const U4*)(vr + hi16);
            bv0.h[1] = *(const U4*)(vr + hi16 + 8);
            bv1.h[0] = *(const U4*)(vr + 32 + hi16);
            bv1.h[1] = *(const U4*)(vr + 32 + hi16 + 8);
            O[c] = WMMA_BF16(pa0.v, bv0.v, O[c]);
            O[c] = WMMA_BF16(pa1.v, bv1.v, O[c]);
        }
        __syncthreads();   // protect kt/vt before next iteration's staging
    }

    // ---- normalize and store attn output as bf16 [b, t, h*e] ------------
    #pragma unroll
    for (int r = 0; r < 8; r++) {
        float inv = 1.0f / lrun[r];
        int row = q0 + hi8 + r;
        size_t base = ((size_t)b * T + row) * NE + h * E;
        #pragma unroll
        for (int c = 0; c < 8; c++)
            aout[base + c * 16 + lm] = f2bf(O[c][r] * inv);
    }
}

// ---------------------------------------------------------------------------
// Kernel 3: output projection.  Y[8192,128] = A[8192,1024] @ Wu + bu, f32 out.
// One wave per 16x32 output strip (2 accumulators, A-fragment reused).
// ---------------------------------------------------------------------------
__global__ __launch_bounds__(128, 2)
void outproj_kernel(const __bf16* __restrict__ aout,
                    const __bf16* __restrict__ WuT,
                    const float*  __restrict__ bu,
                    float*        __restrict__ out) {
    int wave = threadIdx.x >> 5, lane = threadIdx.x & 31;
    int job = blockIdx.x * 4 + wave;        // 512 mtiles * 4 nblocks = 2048
    int m0 = (job >> 2) * 16, n0 = (job & 3) * 32;
    int lm = lane & 15, hi = lane >> 4, hi8 = hi * 8, hi16 = hi * 16;

    v8f acc0 = {}, acc1 = {};
    const __bf16* ar  = aout + (size_t)(m0 + lm) * NE;
    const __bf16* wr0 = WuT  + (size_t)(n0 + lm) * NE;
    const __bf16* wr1 = wr0  + (size_t)16 * NE;
    #pragma unroll 4
    for (int kc = 0; kc < 32; kc++) {
        V16 a, w0, w1;
        a.h[0]  = *(const U4*)(ar + kc * 32 + hi16);
        a.h[1]  = *(const U4*)(ar + kc * 32 + hi16 + 8);
        w0.h[0] = *(const U4*)(wr0 + kc * 32 + hi16);
        w0.h[1] = *(const U4*)(wr0 + kc * 32 + hi16 + 8);
        w1.h[0] = *(const U4*)(wr1 + kc * 32 + hi16);
        w1.h[1] = *(const U4*)(wr1 + kc * 32 + hi16 + 8);
        acc0 = WMMA_BF16(a.v, w0.v, acc0);
        acc1 = WMMA_BF16(a.v, w1.v, acc1);
    }
    float bias0 = bu[n0 + lm], bias1 = bu[n0 + 16 + lm];
    #pragma unroll
    for (int r = 0; r < 8; r++) {
        size_t row = (size_t)(m0 + hi8 + r) * E;
        out[row + n0 + lm]      = acc0[r] + bias0;
        out[row + n0 + 16 + lm] = acc1[r] + bias1;
    }
}

// ---------------------------------------------------------------------------
extern "C" void kernel_launch(void* const* d_in, const int* in_sizes, int n_in,
                              void* d_out, int out_size, void* d_ws, size_t ws_size,
                              hipStream_t stream) {
    const float* x    = (const float*)d_in[0];
    const int*   mask = (const int*)  d_in[1];
    const float* Wq   = (const float*)d_in[2];
    const float* Wk   = (const float*)d_in[3];
    const float* Wv   = (const float*)d_in[4];
    const float* Wu   = (const float*)d_in[5];
    const float* bu   = (const float*)d_in[6];

    // Workspace layout (bf16 elems):
    //   qb,kb,vT,aout : 8 Mi each (64 MB total)
    //   WuT 128 Ki, xb 1 Mi, WT 384 Ki   (~67 MB total; fits global L2)
    __bf16* qb   = (__bf16*)d_ws;
    __bf16* kb   = qb   + (size_t)M * NE;
    __bf16* vT   = kb   + (size_t)M * NE;
    __bf16* aout = vT   + (size_t)M * NE;
    __bf16* WuT  = aout + (size_t)M * NE;
    __bf16* xb   = WuT  + (size_t)E * NE;
    __bf16* WT   = xb   + (size_t)M * E;

    xconv_kernel  <<<4096, 256, 0, stream>>>(x, xb);
    wconv_kernel  <<<1536, 256, 0, stream>>>(Wq, Wk, Wv, WT);
    wuconv_kernel <<<512,  256, 0, stream>>>(Wu, WuT);
    qkv_kernel    <<<3072, 128, 0, stream>>>(xb, WT, qb, kb, vT);
    attn_kernel   <<<1024, 128, 0, stream>>>(qb, kb, vT, mask, aout);
    outproj_kernel<<<512,  128, 0, stream>>>(aout, WuT, bu, (float*)d_out);
}
